// DecoderWithAttention_89567247991543
// MI455X (gfx1250) — compile-verified, matrix operand order checked by reading, not run on hardware
//
#include <hip/hip_runtime.h>
#include <hip/hip_bf16.h>

typedef __bf16 bf16;
typedef __attribute__((ext_vector_type(16))) __bf16 v16bf;
typedef __attribute__((ext_vector_type(8)))  __bf16 v8bf;
typedef __attribute__((ext_vector_type(4)))  __bf16 v4bf;
typedef __attribute__((ext_vector_type(8)))  float  v8f;

// ---------------------------------------------------------------------------
// bf16 fragment: 16 bf16 for one lane of a 16x16x32 bf16 WMMA.
// elements j=0..7 at base[0..7] (16B), j=8..15 at base[16..23] (16B).
// Two global_load_b128, no converts.
// ---------------------------------------------------------------------------
__device__ __forceinline__ v16bf load_bf_frag(const bf16* __restrict__ base) {
  const v8bf* p = (const v8bf*)base;
  const v8bf lo = p[0];
  const v8bf hi = p[2];
  v16bf f;
#pragma unroll
  for (int j = 0; j < 8; ++j) { f[j] = lo[j]; f[8 + j] = hi[j]; }
  return f;
}

// ---------------------------------------------------------------------------
// Pure-bf16 GEMM: C[M,N] = A[M,K] * Bt^T (+ bias).  A: [M,K] bf16 row-major,
// Bt: [N,K] bf16 row-major (i.e. B transposed).  Optional bf16 copy of C.
// Wave computes a 16 x (16*NT) strip: one A fragment reused over NT B tiles.
// M%16==0, N%(16*NT)==0, K%32==0.  blockDim = 128 (4 waves).
// ---------------------------------------------------------------------------
template <int NT>
__global__ void gemm_bf16(const bf16* __restrict__ A, const bf16* __restrict__ Bt,
                          const float* __restrict__ bias, float* __restrict__ C,
                          bf16* __restrict__ Cbf, int M, int N, int K) {
  const int wave  = threadIdx.x >> 5;
  const int lane  = threadIdx.x & 31;
  const int group = blockIdx.x * (blockDim.x >> 5) + wave;
  const int ngn   = N / (16 * NT);
  const int tm    = group / ngn;
  const int gn    = group - tm * ngn;
  const int half  = lane >> 4;
  const int l15   = lane & 15;
  const int arow  = tm * 16 + l15;
  const int ncol0 = gn * (16 * NT) + l15;

  v8f acc[NT] = {};
  for (int k0 = 0; k0 < K; k0 += 32) {
    const v16bf af = load_bf_frag(A + (size_t)arow * K + k0 + half * 8);
#pragma unroll
    for (int u = 0; u < NT; ++u) {
      const v16bf bfrag =
          load_bf_frag(Bt + (size_t)(ncol0 + u * 16) * K + k0 + half * 8);
      acc[u] = __builtin_amdgcn_wmma_f32_16x16x32_bf16(false, af, false, bfrag,
                                                       (short)0, acc[u], false, false);
    }
  }

#pragma unroll
  for (int u = 0; u < NT; ++u) {
    const int bcol = ncol0 + u * 16;
    const float bb = bias[bcol];
#pragma unroll
    for (int r = 0; r < 8; ++r) {
      const int row = tm * 16 + half * 8 + r;
      const float v = acc[u][r] + bb;
      C[(size_t)row * N + bcol] = v;
      if (Cbf != nullptr) Cbf[(size_t)row * N + bcol] = (bf16)v;
    }
  }
}

// ---------------------------------------------------------------------------
// Final FC: preds[b,t,:] = mask ? h @ W_fc + b_fc : 0   (M=64,K=512,N=10000)
// A: h bf16 [64,512]; Wt: W_fc^T bf16 [10000,512].  NT=5 (10000 = 80*125).
// ---------------------------------------------------------------------------
__global__ void fc_wmma(const bf16* __restrict__ H, const bf16* __restrict__ Wt,
                        const float* __restrict__ bfc, float* __restrict__ outp,
                        const int* __restrict__ dec_len, int t, int T) {
  const int K = 512, N = 10000, NT = 5;
  const int wave  = threadIdx.x >> 5;
  const int lane  = threadIdx.x & 31;
  const int group = blockIdx.x * (blockDim.x >> 5) + wave;
  const int ngn   = N / (16 * NT);      // 125
  const int tm    = group / ngn;
  const int gn    = group - tm * ngn;
  const int half  = lane >> 4;
  const int l15   = lane & 15;
  const int arow  = tm * 16 + l15;
  const int ncol0 = gn * (16 * NT) + l15;

  v8f acc[NT] = {};
  for (int k0 = 0; k0 < K; k0 += 32) {
    const v16bf af = load_bf_frag(H + (size_t)arow * K + k0 + half * 8);
#pragma unroll
    for (int u = 0; u < NT; ++u) {
      const v16bf bfrag =
          load_bf_frag(Wt + (size_t)(ncol0 + u * 16) * K + k0 + half * 8);
      acc[u] = __builtin_amdgcn_wmma_f32_16x16x32_bf16(false, af, false, bfrag,
                                                       (short)0, acc[u], false, false);
    }
  }

  int dls[8];
#pragma unroll
  for (int r = 0; r < 8; ++r) dls[r] = dec_len[tm * 16 + half * 8 + r];

#pragma unroll
  for (int u = 0; u < NT; ++u) {
    const int bcol = ncol0 + u * 16;
    const float bb = bfc[bcol];
#pragma unroll
    for (int r = 0; r < 8; ++r) {
      const int row = tm * 16 + half * 8 + r;           // b in [0,64)
      const float v = acc[u][r] + bb;
      outp[((size_t)row * T + t) * (size_t)N + bcol] = (dls[r] > t) ? v : 0.f;
    }
  }
}

// ---------------------------------------------------------------------------
// Weight prep: f32 -> bf16, straight copy (4 elements per thread).
// ---------------------------------------------------------------------------
__global__ void f32_to_bf16_k(const float* __restrict__ in, bf16* __restrict__ out) {
  const int i = blockIdx.x * blockDim.x + threadIdx.x;
  const float4 v = ((const float4*)in)[i];
  v4bf o;
  o[0] = (bf16)v.x; o[1] = (bf16)v.y; o[2] = (bf16)v.z; o[3] = (bf16)v.w;
  ((v4bf*)out)[i] = o;
}

// Weight prep: f32 [K,N] -> bf16 [N,K] transpose (coalesced writes).
__global__ void transpose_bf16_k(const float* __restrict__ in, bf16* __restrict__ out,
                                 int K, int N) {
  const int idx = blockIdx.x * blockDim.x + threadIdx.x;  // over N*K
  const int n = idx / K;
  const int k = idx - n * K;
  out[idx] = (bf16)in[(size_t)k * N + n];
}

// ---------------------------------------------------------------------------
// Sort batch by caption length (descending, stable) + emit caps/declen/sortind.
// ---------------------------------------------------------------------------
__global__ void sort_init(const int* __restrict__ cap_len, const int* __restrict__ caps,
                          int T, int* __restrict__ sort_ind, int* __restrict__ dec_len,
                          float* __restrict__ out_caps, float* __restrict__ out_declen,
                          float* __restrict__ out_sortind) {
  __shared__ int len_sh[64];
  __shared__ int ord[64];
  const int i = threadIdx.x;
  len_sh[i] = cap_len[i];
  __syncthreads();
  const int li = len_sh[i];
  int rank = 0;
  for (int j = 0; j < 64; ++j) {
    const int lj = len_sh[j];
    if (lj > li || (lj == li && j < i)) rank++;
  }
  ord[rank] = i;
  __syncthreads();
  const int s = ord[i];
  sort_ind[i] = s;
  const int dl = len_sh[s] - 1;
  dec_len[i] = dl;
  out_sortind[i] = (float)s;
  out_declen[i] = (float)dl;
  for (int t = 0; t < T; ++t) out_caps[i * T + t] = (float)caps[s * T + t];
}

// enc_s[b] = encoder_out[sort_ind[b]] (f32) + bf16 copy for the att1 GEMM.
__global__ void gather_enc(const float* __restrict__ enc, const int* __restrict__ sort_ind,
                           float* __restrict__ enc_s, bf16* __restrict__ enc_bf) {
  const size_t idx = (size_t)blockIdx.x * blockDim.x + threadIdx.x;
  const int b = (int)(idx / (196u * 1024u));
  const size_t rem = idx - (size_t)b * 196 * 1024;
  const float v = enc[(size_t)sort_ind[b] * 196 * 1024 + rem];
  enc_s[idx] = v;
  enc_bf[idx] = (bf16)v;
}

// mean over P=196 -> bf16 (feeds h0 GEMM only)
__global__ void mean_enc_k(const float* __restrict__ enc_s, bf16* __restrict__ mean_bf) {
  const int idx = blockIdx.x * blockDim.x + threadIdx.x;  // 64*1024
  const int b = idx >> 10, d = idx & 1023;
  const float* p = enc_s + (size_t)b * 196 * 1024 + d;
  float s = 0.f;
  for (int i = 0; i < 196; ++i) s += p[(size_t)i * 1024];
  mean_bf[idx] = (bf16)(s * (1.f / 196.f));
}

// ---------------------------------------------------------------------------
// Fused attention step: e = relu(att1 + att2) @ w_full; softmax over P;
// awe = alpha @ enc; gate = sigmoid(gatepre); x = [emb_t | gate*awe] (bf16);
// masked alphas written to output.  One block (256 threads) per batch row.
// ---------------------------------------------------------------------------
__global__ void attn_step(const float* __restrict__ att1, const float* __restrict__ att2,
                          const float* __restrict__ w_full, const float* __restrict__ b_full,
                          const float* __restrict__ enc_s, const float* __restrict__ gatepre,
                          const float* __restrict__ emb, const int* __restrict__ caps,
                          const int* __restrict__ sort_ind, const int* __restrict__ dec_len,
                          bf16* __restrict__ xbf, float* __restrict__ alphas_out,
                          int t, int T) {
  __shared__ float e_sh[196];
  __shared__ float red[8];
  const int b = blockIdx.x;
  const int tid = threadIdx.x;
  const int lane = tid & 31, wid = tid >> 5;

  // e scores (float4 over A=512)
  for (int p = tid; p < 196; p += 256) {
    const float4* a1 = (const float4*)(att1 + ((size_t)b * 196 + p) * 512);
    const float4* a2 = (const float4*)(att2 + (size_t)b * 512);
    const float4* wf = (const float4*)w_full;
    float acc = 0.f;
    for (int q = 0; q < 128; ++q) {
      const float4 x1 = a1[q], x2 = a2[q], ww = wf[q];
      acc = fmaf(fmaxf(x1.x + x2.x, 0.f), ww.x, acc);
      acc = fmaf(fmaxf(x1.y + x2.y, 0.f), ww.y, acc);
      acc = fmaf(fmaxf(x1.z + x2.z, 0.f), ww.z, acc);
      acc = fmaf(fmaxf(x1.w + x2.w, 0.f), ww.w, acc);
    }
    e_sh[p] = acc + b_full[0];
  }
  __syncthreads();

  // softmax max
  float m = -3.4e38f;
  for (int p = tid; p < 196; p += 256) m = fmaxf(m, e_sh[p]);
  for (int off = 16; off > 0; off >>= 1) m = fmaxf(m, __shfl_xor(m, off, 32));
  if (lane == 0) red[wid] = m;
  __syncthreads();
  m = red[0];
  for (int i = 1; i < 8; ++i) m = fmaxf(m, red[i]);
  __syncthreads();

  // exp + sum
  float s = 0.f;
  for (int p = tid; p < 196; p += 256) {
    const float ev = __expf(e_sh[p] - m);
    e_sh[p] = ev;
    s += ev;
  }
  for (int off = 16; off > 0; off >>= 1) s += __shfl_xor(s, off, 32);
  if (lane == 0) red[wid] = s;
  __syncthreads();
  s = 0.f;
  for (int i = 0; i < 8; ++i) s += red[i];
  const float inv = 1.f / s;
  const int mask = dec_len[b] > t;
  for (int p = tid; p < 196; p += 256) {
    const float al = e_sh[p] * inv;
    e_sh[p] = al;
    alphas_out[((size_t)b * T + t) * 196 + p] = mask ? al : 0.f;
  }
  __syncthreads();

  // gated attention-weighted encoding -> x[:, 512:1536] (bf16), float4 lanes
  {
    const int d0 = tid * 4;  // 256 threads cover 1024
    const float4* ep = (const float4*)(enc_s + (size_t)b * 196 * 1024 + d0);
    float4 a = {0.f, 0.f, 0.f, 0.f};
    for (int p = 0; p < 196; ++p) {
      const float al = e_sh[p];
      const float4 ev = ep[p * 256];  // row stride 1024 floats
      a.x = fmaf(al, ev.x, a.x);
      a.y = fmaf(al, ev.y, a.y);
      a.z = fmaf(al, ev.z, a.z);
      a.w = fmaf(al, ev.w, a.w);
    }
    const float4 gp = *(const float4*)(gatepre + (size_t)b * 1024 + d0);
    v4bf o;
    o[0] = (bf16)(a.x / (1.f + __expf(-gp.x)));
    o[1] = (bf16)(a.y / (1.f + __expf(-gp.y)));
    o[2] = (bf16)(a.z / (1.f + __expf(-gp.z)));
    o[3] = (bf16)(a.w / (1.f + __expf(-gp.w)));
    *(v4bf*)(xbf + (size_t)b * 1536 + 512 + d0) = o;
  }

  // embedding -> x[:, 0:512] (bf16)
  const int tok = caps[(size_t)sort_ind[b] * T + t];
  if (tid < 128) {
    const float4 e4 = ((const float4*)(emb + (size_t)tok * 512))[tid];
    v4bf o;
    o[0] = (bf16)e4.x; o[1] = (bf16)e4.y; o[2] = (bf16)e4.z; o[3] = (bf16)e4.w;
    ((v4bf*)(xbf + (size_t)b * 1536))[tid] = o;
  }
}

// ---------------------------------------------------------------------------
// GRU pointwise update with row mask.  gi/gh already include biases.
// Writes f32 h (for next-step reuse) and bf16 h (for GEMM consumers).
// ---------------------------------------------------------------------------
__global__ void gru_step(const float* __restrict__ gi, const float* __restrict__ gh,
                         const float* __restrict__ hin, const int* __restrict__ dec_len,
                         int t, float* __restrict__ hout, bf16* __restrict__ houtbf) {
  const int idx = blockIdx.x * blockDim.x + threadIdx.x;  // 64*512
  const int b = idx >> 9, j = idx & 511;
  const size_t g = (size_t)b * 1536 + j;
  const float i_r = gi[g], i_z = gi[g + 512], i_n = gi[g + 1024];
  const float h_r = gh[g], h_z = gh[g + 512], h_n = gh[g + 1024];
  const float r = 1.f / (1.f + __expf(-(i_r + h_r)));
  const float z = 1.f / (1.f + __expf(-(i_z + h_z)));
  const float n = tanhf(i_n + r * h_n);
  const float hprev = hin[(size_t)b * 512 + j];
  const float hnew = (1.f - z) * n + z * hprev;
  const float hv = (dec_len[b] > t) ? hnew : hprev;
  hout[(size_t)b * 512 + j] = hv;
  houtbf[(size_t)b * 512 + j] = (bf16)hv;
}

// ---------------------------------------------------------------------------
extern "C" void kernel_launch(void* const* d_in, const int* in_sizes, int n_in,
                              void* d_out, int out_size, void* d_ws, size_t ws_size,
                              hipStream_t stream) {
  const float* encoder_out = (const float*)d_in[0];
  const int*   caps        = (const int*)d_in[1];
  const int*   cap_len     = (const int*)d_in[2];
  const float* emb         = (const float*)d_in[4];
  const float* W_enc_att   = (const float*)d_in[5];
  const float* b_enc_att   = (const float*)d_in[6];
  const float* W_dec_att   = (const float*)d_in[7];
  const float* b_dec_att   = (const float*)d_in[8];
  const float* w_full      = (const float*)d_in[9];
  const float* b_full      = (const float*)d_in[10];
  const float* W_init_h    = (const float*)d_in[11];
  const float* b_init_h    = (const float*)d_in[12];
  const float* W_fbeta     = (const float*)d_in[13];
  const float* b_fbeta     = (const float*)d_in[14];
  const float* W_ih        = (const float*)d_in[15];
  const float* W_hh        = (const float*)d_in[16];
  const float* b_ih        = (const float*)d_in[17];
  const float* b_hh        = (const float*)d_in[18];
  const float* W_fc        = (const float*)d_in[19];
  const float* b_fc        = (const float*)d_in[20];

  const int T = in_sizes[1] / 64;  // 52

  // ---- output layout (return order, flattened) ----
  float* out = (float*)d_out;
  float* out_pred    = out;
  float* out_caps    = out_pred + (size_t)64 * T * 10000;
  float* out_declen  = out_caps + (size_t)64 * T;
  float* out_alphas  = out_declen + 64;
  float* out_sortind = out_alphas + (size_t)64 * T * 196;

  // ---- workspace layout ----
  char* ws = (char*)d_ws;
  size_t off = 0;
  auto alloc = [&](size_t bytes) {
    size_t r = off;
    off = (off + bytes + 255) & ~(size_t)255;
    return r;
  };
  float* enc_s    = (float*)(ws + alloc((size_t)64 * 196 * 1024 * 4));
  bf16*  enc_bf   = (bf16*)(ws + alloc((size_t)64 * 196 * 1024 * 2));
  float* att1     = (float*)(ws + alloc((size_t)64 * 196 * 512 * 4));
  bf16*  Wenc_t   = (bf16*)(ws + alloc((size_t)512 * 1024 * 2));    // [512,1024]
  bf16*  Wdec_t   = (bf16*)(ws + alloc((size_t)512 * 512 * 2));     // [512,512]
  bf16*  Wfbeta_t = (bf16*)(ws + alloc((size_t)1024 * 512 * 2));    // [1024,512]
  bf16*  Winit_t  = (bf16*)(ws + alloc((size_t)512 * 1024 * 2));    // [512,1024]
  bf16*  Wih_bf   = (bf16*)(ws + alloc((size_t)1536 * 1536 * 2));   // [1536,1536]
  bf16*  Whh_bf   = (bf16*)(ws + alloc((size_t)1536 * 512 * 2));    // [1536,512]
  bf16*  Wfc_t    = (bf16*)(ws + alloc((size_t)10000 * 512 * 2));   // [10000,512]
  float* att2     = (float*)(ws + alloc((size_t)64 * 512 * 4));
  float* gatepre  = (float*)(ws + alloc((size_t)64 * 1024 * 4));
  bf16*  x_bf     = (bf16*)(ws + alloc((size_t)64 * 1536 * 2));
  float* gi       = (float*)(ws + alloc((size_t)64 * 1536 * 4));
  float* gh       = (float*)(ws + alloc((size_t)64 * 1536 * 4));
  float* hA       = (float*)(ws + alloc((size_t)64 * 512 * 4));
  float* hB       = (float*)(ws + alloc((size_t)64 * 512 * 4));
  bf16*  hAbf     = (bf16*)(ws + alloc((size_t)64 * 512 * 2));
  bf16*  hBbf     = (bf16*)(ws + alloc((size_t)64 * 512 * 2));
  bf16*  mean_bf  = (bf16*)(ws + alloc((size_t)64 * 1024 * 2));
  int*   sort_ind = (int*)(ws + alloc(64 * 4));
  int*   dec_len  = (int*)(ws + alloc(64 * 4));

  // ---- weight prep (bf16, GEMM-friendly [N,K] layouts) ----
  transpose_bf16_k<<<(512 * 1024) / 256, 256, 0, stream>>>(W_enc_att, Wenc_t, 1024, 512);
  transpose_bf16_k<<<(512 * 512) / 256, 256, 0, stream>>>(W_dec_att, Wdec_t, 512, 512);
  transpose_bf16_k<<<(1024 * 512) / 256, 256, 0, stream>>>(W_fbeta, Wfbeta_t, 512, 1024);
  transpose_bf16_k<<<(512 * 1024) / 256, 256, 0, stream>>>(W_init_h, Winit_t, 1024, 512);
  transpose_bf16_k<<<(10000 * 512) / 256, 256, 0, stream>>>(W_fc, Wfc_t, 512, 10000);
  f32_to_bf16_k<<<(1536 * 1536) / 1024, 256, 0, stream>>>(W_ih, Wih_bf);
  f32_to_bf16_k<<<(1536 * 512) / 1024, 256, 0, stream>>>(W_hh, Whh_bf);

  // ---- setup ----
  sort_init<<<1, 64, 0, stream>>>(cap_len, caps, T, sort_ind, dec_len,
                                  out_caps, out_declen, out_sortind);
  gather_enc<<<(64 * 196 * 1024) / 256, 256, 0, stream>>>(encoder_out, sort_ind,
                                                          enc_s, enc_bf);
  mean_enc_k<<<(64 * 1024) / 256, 256, 0, stream>>>(enc_s, mean_bf);
  // att1[12544,512] = enc @ W_enc_att + b_enc_att
  gemm_bf16<4><<<(784 * 8) / 4, 128, 0, stream>>>(enc_bf, Wenc_t, b_enc_att, att1,
                                                  nullptr, 12544, 512, 1024);
  // h0[64,512] = mean_enc @ W_init_h + b_init_h  (f32 + bf16 copies)
  gemm_bf16<4><<<(4 * 8) / 4, 128, 0, stream>>>(mean_bf, Winit_t, b_init_h, hA,
                                                hAbf, 64, 512, 1024);

  // ---- time steps ----
  for (int t = 0; t < T; ++t) {
    const float* hin   = (t & 1) ? hB : hA;
    const bf16*  hinbf = (t & 1) ? hBbf : hAbf;
    float*       hout   = (t & 1) ? hA : hB;
    bf16*        houtbf = (t & 1) ? hAbf : hBbf;

    // att2 = h @ W_dec_att + b_dec_att          [64,512]
    gemm_bf16<4><<<(4 * 8) / 4, 128, 0, stream>>>(hinbf, Wdec_t, b_dec_att, att2,
                                                  nullptr, 64, 512, 512);
    // gatepre = h @ W_fbeta + b_fbeta           [64,1024]
    gemm_bf16<4><<<(4 * 16) / 4, 128, 0, stream>>>(hinbf, Wfbeta_t, b_fbeta, gatepre,
                                                   nullptr, 64, 1024, 512);
    // gh = h @ W_hh.T + b_hh                    [64,1536]
    gemm_bf16<4><<<(4 * 24) / 4, 128, 0, stream>>>(hinbf, Whh_bf, b_hh, gh,
                                                   nullptr, 64, 1536, 512);
    // attention + gate + x assembly (bf16 x)
    attn_step<<<64, 256, 0, stream>>>(att1, att2, w_full, b_full, enc_s, gatepre,
                                      emb, caps, sort_ind, dec_len, x_bf,
                                      out_alphas, t, T);
    // gi = x @ W_ih.T + b_ih                    [64,1536]
    gemm_bf16<4><<<(4 * 24) / 4, 128, 0, stream>>>(x_bf, Wih_bf, b_ih, gi,
                                                   nullptr, 64, 1536, 1536);
    // GRU pointwise + mask (f32 + bf16 h)
    gru_step<<<(64 * 512) / 256, 256, 0, stream>>>(gi, gh, hin, dec_len, t,
                                                   hout, houtbf);
    // preds = h_next @ W_fc + b_fc (masked)     [64,10000]
    fc_wmma<<<(4 * 125) / 4, 128, 0, stream>>>(houtbf, Wfc_t, b_fc, out_pred,
                                               dec_len, t, T);
  }
}
